// RandNLAGQALayer_13048110645529
// MI455X (gfx1250) — matrix operand order repaired and checked by reading, not compiled, not run on hardware
//
#include <hip/hip_runtime.h>
#include <math.h>

// ---------------------------------------------------------------------------
// RandNLA GQA layer for MI455X (gfx1250), wave32 + WMMA bf16.
// B=2, S=8192, HID=2048, H=16, HKV=4, D=128, SK=640.
// ---------------------------------------------------------------------------

typedef __bf16 bf16_t;
typedef __attribute__((ext_vector_type(16))) __bf16 v16bf;
typedef __attribute__((ext_vector_type(8)))  __bf16 v8bf;
typedef __attribute__((ext_vector_type(8)))  float  v8f;

#define BB   2
#define SS   8192
#define HID  2048
#define NH   16
#define NKV  4
#define DD   128
#define SK   640
#define NTOK (BB * SS)

// ---------------------------------------------------------------------------
// Fragment gather: 16 bf16 per lane as two 16-byte chunks `step` apart.
// A-matrix 16x32 bf16: lane L<16 -> row=L, K=0..15 contig (same order as B) ->
//   base = row*lda + (hi?16:0), step=8.
// B-matrix 32x16 bf16 (N-major source): lane L<16 -> col=L, K=0..15 contig ->
//   base = col*ldb + (hi?16:0), step=8.
// ---------------------------------------------------------------------------
__device__ __forceinline__ v16bf pack16(const bf16_t* p, int step) {
  v8bf lo = *(const v8bf*)(p);
  v8bf hv = *(const v8bf*)(p + step);
  v16bf f;
#pragma unroll
  for (int i = 0; i < 8; ++i) { f[i] = lo[i]; f[8 + i] = hv[i]; }
  return f;
}

// One wave computes a 16(M) x 128(N) strip: acc[8] of 16x16 f32 C-fragments.
// A: row-major MxK bf16.  BT: N-major (N x K) bf16.  K multiple of 64.
// Ping-pong double buffering (2x unrolled): loads write directly into the
// registers their WMMA consumes, so no rotation copies; each 8-WMMA chain
// hides the other phase's 18 global_load_b128.
// NOTE: the final phase prefetches ~64B past K; callers guarantee the operand
// buffer is followed by more workspace, so the dead over-read is safe.
__device__ __forceinline__ void wave_gemm_strip(const bf16_t* A, long lda, int row0,
                                                const bf16_t* BT, long ldb, int col0,
                                                int K, v8f acc[8]) {
  const int lane = threadIdx.x & 31;
  const int lo16 = lane & 15, hi = lane >> 4;
  const bf16_t* pA = A + (long)(row0 + lo16) * lda + hi * 16;
  const bf16_t* pB[8];
#pragma unroll
  for (int nf = 0; nf < 8; ++nf)
    pB[nf] = BT + (long)(col0 + nf * 16 + lo16) * ldb + hi * 16;

  v16bf a0 = pack16(pA, 8);
  v16bf b0[8];
#pragma unroll
  for (int nf = 0; nf < 8; ++nf) b0[nf] = pack16(pB[nf], 8);

  for (int k0 = 0; k0 < K; k0 += 64) {
    // phase 1 operands (k0+32) into ping buffers
    __builtin_prefetch(pA + k0 + 256, 0, 1);
    v16bf a1 = pack16(pA + k0 + 32, 8);
    v16bf b1[8];
#pragma unroll
    for (int nf = 0; nf < 8; ++nf) b1[nf] = pack16(pB[nf] + k0 + 32, 8);
    // compute phase 0
#pragma unroll
    for (int nf = 0; nf < 8; ++nf)
      acc[nf] = __builtin_amdgcn_wmma_f32_16x16x32_bf16(
          false, a0, false, b0[nf], (short)0, acc[nf], false, false);
    // phase 0 operands for next trip (k0+64; dead over-read on last trip)
    a0 = pack16(pA + k0 + 64, 8);
#pragma unroll
    for (int nf = 0; nf < 8; ++nf) b0[nf] = pack16(pB[nf] + k0 + 64, 8);
    // compute phase 1
#pragma unroll
    for (int nf = 0; nf < 8; ++nf)
      acc[nf] = __builtin_amdgcn_wmma_f32_16x16x32_bf16(
          false, a1, false, b1[nf], (short)0, acc[nf], false, false);
  }
}

// ---------------------------------------------------------------------------
// Kernel 1: importance sigmoid per token + x -> bf16.  One block per token.
// ---------------------------------------------------------------------------
__global__ __launch_bounds__(256) void imp_x_kernel(
    const float* __restrict__ x, const float* __restrict__ w1,
    const float* __restrict__ b1, const float* __restrict__ w2,
    const float* __restrict__ b2, float* __restrict__ iw,
    bf16_t* __restrict__ xbf) {
  __shared__ float xs[HID];
  __shared__ float red[256];
  const int t   = blockIdx.x;
  const int tid = threadIdx.x;
  const long xoff = (long)t * HID;
#pragma unroll
  for (int e = 0; e < HID / 256; ++e) xs[e * 256 + tid] = x[xoff + e * 256 + tid];
  __syncthreads();

  const int j = tid & 63, quad = tid >> 6;       // 4 partial sums per hidden j
  float s = 0.f;
  for (int i = quad * 512; i < quad * 512 + 512; ++i) s += xs[i] * w1[i * 64 + j];
  red[tid] = s;
  __syncthreads();
  if (tid < 64) {
    float h = tanhf(red[tid] + red[tid + 64] + red[tid + 128] + red[tid + 192] + b1[tid]);
    red[tid] = h * w2[tid];
  }
  __syncthreads();
  if (tid == 0) {
    float acc = 0.f;
#pragma unroll
    for (int i = 0; i < 64; ++i) acc += red[i];
    float logit = acc + b2[0] - logf((float)SS / (float)SK);  // pressure bias
    iw[t] = 1.f / (1.f + expf(-logit));
  }
#pragma unroll
  for (int e = 0; e < HID / 256; ++e)
    xbf[xoff + e * 256 + tid] = (bf16_t)xs[e * 256 + tid];
}

// ---------------------------------------------------------------------------
// Kernel 2: build transposed bf16 QKV weight  wqkvT[n][k], n in [0,3072)
// ---------------------------------------------------------------------------
__global__ __launch_bounds__(256) void build_wqkvT(
    const float* __restrict__ wq, const float* __restrict__ wk,
    const float* __restrict__ wv, bf16_t* __restrict__ wt) {
  const long gid = (long)blockIdx.x * 256 + threadIdx.x;  // over 3072*2048
  const int n = (int)(gid >> 11), k = (int)(gid & 2047);
  float v;
  if (n < 2048)      v = wq[(long)k * 2048 + n];
  else if (n < 2560) v = wk[(long)k * 512 + (n - 2048)];
  else               v = wv[(long)k * 512 + (n - 2560)];
  wt[gid] = (bf16_t)v;
}

__global__ __launch_bounds__(256) void build_woT(
    const float* __restrict__ wo, bf16_t* __restrict__ wt) {
  const long gid = (long)blockIdx.x * 256 + threadIdx.x;  // over 2048*2048
  const int n = (int)(gid >> 11), k = (int)(gid & 2047);
  wt[gid] = (bf16_t)wo[(long)k * 2048 + n];
}

// Kernel 3: Kronecker sketch matrix  P[m][s] = a[m/32, s/256] * b[m%32, s%256]
__global__ __launch_bounds__(256) void build_P(
    const float* __restrict__ ka, const float* __restrict__ kb,
    bf16_t* __restrict__ P) {
  const long gid = (long)blockIdx.x * 256 + threadIdx.x;  // over 640*8192
  const int m = (int)(gid >> 13), s = (int)(gid & 8191);
  P[gid] = (bf16_t)(ka[(m >> 5) * 128 + (s >> 8)] * kb[(m & 31) * 256 + (s & 255)]);
}

// ---------------------------------------------------------------------------
// Kernel 4: fused QKV GEMM.  C(16384 x 3072) tiles of 128x128 = one head each.
// Epilogue: q -> RMS+RoPE -> q_bf16; k -> RMS*iw; v -> *iw; k/v written
// sequence-transposed into kvwT[b][col(1024)][s(8192)].
// ---------------------------------------------------------------------------
__global__ __launch_bounds__(256) void qkv_gemm(
    const bf16_t* __restrict__ xbf, const bf16_t* __restrict__ wqkvT,
    const float* __restrict__ cosp, const float* __restrict__ sinp,
    const float* __restrict__ qnw, const float* __restrict__ knw,
    const float* __restrict__ iw, bf16_t* __restrict__ qbf,
    bf16_t* __restrict__ kvwT) {
  const int lane = threadIdx.x & 31, wave = threadIdx.x >> 5;
  const int row0 = blockIdx.x * 128 + wave * 16;
  const int colT = blockIdx.y;            // 0..15 q heads, 16..19 k, 20..23 v
  v8f acc[8] = {};
  wave_gemm_strip(xbf, HID, row0, wqkvT, HID, colT * 128, HID, acc);

  const int nloc = lane & 15, hi = lane >> 4;
  // per-row RMS denominators over the 128 cols of this tile (one head)
  float invr[8];
#pragma unroll
  for (int r = 0; r < 8; ++r) {
    float rs = 0.f;
#pragma unroll
    for (int nf = 0; nf < 8; ++nf) rs += acc[nf][r] * acc[nf][r];
#pragma unroll
    for (int m = 1; m < 16; m <<= 1) rs += __shfl_xor(rs, m, 32);
    // all lanes in each 16-lane half now hold the sum for row (r + 8*hi)
    invr[r] = rsqrtf(rs * (1.f / DD) + 1e-6f);
  }

#pragma unroll
  for (int r = 0; r < 8; ++r) {
    const int t = row0 + r + 8 * hi;
    const int s = t & (SS - 1);
    const int b = t >> 13;
    const float wgt = iw[t];
#pragma unroll
    for (int nf = 0; nf < 8; ++nf) {
      const int d = nf * 16 + nloc;
      if (colT < 16) {                      // ---- Q: RMS + RoPE ----
        float qn = acc[nf][r] * invr[r] * qnw[d];
        int dp = (nf < 4) ? d + 64 : d - 64;
        float qp = acc[nf ^ 4][r] * invr[r] * qnw[dp];
        float rot = (nf < 4) ? -qp : qp;
        float val = qn * cosp[(long)s * DD + d] + rot * sinp[(long)s * DD + d];
        qbf[(long)t * HID + colT * 128 + d] = (bf16_t)val;
      } else if (colT < 20) {               // ---- K: RMS * iw ----
        const int hk = colT - 16;
        float kn = acc[nf][r] * invr[r] * knw[d] * wgt;
        kvwT[((long)b * 1024 + hk * 128 + d) * SS + s] = (bf16_t)kn;
      } else {                              // ---- V: * iw ----
        const int hv = colT - 20;
        float vv = acc[nf][r] * wgt;
        kvwT[((long)b * 1024 + 512 + hv * 128 + d) * SS + s] = (bf16_t)vv;
      }
    }
  }
}

// ---------------------------------------------------------------------------
// Kernel 5: sketch GEMM  C_b(640 x 1024) = P(640x8192) @ KVW_b.
// Epilogue: k part gets RoPE at position m -> k_pe[b][m][hkv][d];
// v part stored transposed -> v_fT[b][hkv][d][m].
// ---------------------------------------------------------------------------
__global__ __launch_bounds__(256) void sketch_gemm(
    const bf16_t* __restrict__ P, const bf16_t* __restrict__ kvwT,
    const float* __restrict__ cosp, const float* __restrict__ sinp,
    bf16_t* __restrict__ kpe, bf16_t* __restrict__ vfT) {
  const int lane = threadIdx.x & 31, wave = threadIdx.x >> 5;
  const int row0 = blockIdx.x * 128 + wave * 16;  // sketch index m strip
  const int colT = blockIdx.y;                    // 0..3 k heads, 4..7 v heads
  const int b    = blockIdx.z;
  v8f acc[8] = {};
  wave_gemm_strip(P, SS, row0, kvwT + (long)b * 1024 * SS, SS, colT * 128, SS, acc);

  const int nloc = lane & 15, hi = lane >> 4;
#pragma unroll
  for (int r = 0; r < 8; ++r) {
    const int m = row0 + r + 8 * hi;
#pragma unroll
    for (int nf = 0; nf < 8; ++nf) {
      const int d = nf * 16 + nloc;
      if (colT < 4) {                        // ---- K sketch: RoPE(m) ----
        float kn  = acc[nf][r];
        float kp  = acc[nf ^ 4][r];
        float rot = (nf < 4) ? -kp : kp;
        float val = kn * cosp[(long)m * DD + d] + rot * sinp[(long)m * DD + d];
        kpe[(((long)b * SK + m) * NKV + colT) * DD + d] = (bf16_t)val;
      } else {                               // ---- V sketch: transpose ----
        const int hv = colT - 4;
        vfT[(((long)b * NKV + hv) * DD + d) * SK + m] = (bf16_t)acc[nf][r];
      }
    }
  }
}

// ---------------------------------------------------------------------------
// Kernel 6: attention.  Block = 4 waves x 16 q-rows, full 640 keys per wave.
// Scores staged in dynamic LDS (64 x 640 f32 = 160 KB), two-pass softmax,
// then P@V with bf16 A-fragments rebuilt from LDS.  K/V fragment loads use
// 2x-unrolled ping-pong buffers against the WMMA chains.
// ---------------------------------------------------------------------------
__global__ __launch_bounds__(128) void attn_kernel(
    const bf16_t* __restrict__ qbf, const bf16_t* __restrict__ kpe,
    const bf16_t* __restrict__ vfT, bf16_t* __restrict__ attnbf) {
  extern __shared__ float sc[];                   // [64][640]
  const int lane = threadIdx.x & 31, wave = threadIdx.x >> 5;
  const int h  = blockIdx.y, b = blockIdx.z;
  const int hk = h >> 2;                          // GQA: n_rep = 4
  const int qrow0 = blockIdx.x * 64 + wave * 16;  // within sequence
  const int nloc = lane & 15, hi = lane >> 4;

  // Q A-fragments for the 4 K-chunks of D=128
  const bf16_t* pQ = qbf + (long)(b * SS + qrow0 + nloc) * HID + h * 128 + hi * 16;
  v16bf qa[4];
#pragma unroll
  for (int c = 0; c < 4; ++c) qa[c] = pack16(pQ + c * 32, 8);

  // ---- scores = Q @ K^T over 40 key-chunks of 16 (ping-pong, 2x unroll) ----
  const bf16_t* pK = kpe + ((long)b * SK + nloc) * (NKV * DD) + hk * DD + hi * 16;
  const long kstride = 16 * (NKV * DD);           // 16 keys
  v16bf kb0[4];
#pragma unroll
  for (int ks = 0; ks < 4; ++ks) kb0[ks] = pack16(pK + ks * 32, 8);
  for (int kc = 0; kc < 40; kc += 2) {
    v16bf kb1[4];
#pragma unroll
    for (int ks = 0; ks < 4; ++ks)
      kb1[ks] = pack16(pK + (kc + 1) * kstride + ks * 32, 8);
    v8f s8 = {};
#pragma unroll
    for (int ks = 0; ks < 4; ++ks)
      s8 = __builtin_amdgcn_wmma_f32_16x16x32_bf16(
          false, qa[ks], false, kb0[ks], (short)0, s8, false, false);
#pragma unroll
    for (int r = 0; r < 8; ++r)
      sc[(wave * 16 + r + 8 * hi) * SK + kc * 16 + nloc] = s8[r];
    // next even chunk (dead over-read into adjacent workspace at kc=38)
#pragma unroll
    for (int ks = 0; ks < 4; ++ks)
      kb0[ks] = pack16(pK + (kc + 2) * kstride + ks * 32, 8);
    v8f s9 = {};
#pragma unroll
    for (int ks = 0; ks < 4; ++ks)
      s9 = __builtin_amdgcn_wmma_f32_16x16x32_bf16(
          false, qa[ks], false, kb1[ks], (short)0, s9, false, false);
#pragma unroll
    for (int r = 0; r < 8; ++r)
      sc[(wave * 16 + r + 8 * hi) * SK + (kc + 1) * 16 + nloc] = s9[r];
  }
  __syncthreads();

  // ---- softmax: 2 threads per row ----
  {
    const int row = threadIdx.x >> 1, half = threadIdx.x & 1;
    float* p = sc + row * SK + half * 320;
    const float scale = 0.08838834764831845f;     // 1/sqrt(128)
    float mx = -3.4e38f;
    for (int i = 0; i < 320; ++i) { float v = p[i] * scale; p[i] = v; mx = fmaxf(mx, v); }
    mx = fmaxf(mx, __shfl_xor(mx, 1, 32));
    float sum = 0.f;
    for (int i = 0; i < 320; ++i) { float e = expf(p[i] - mx); p[i] = e; sum += e; }
    sum += __shfl_xor(sum, 1, 32);
    const float inv = 1.f / sum;
    for (int i = 0; i < 320; ++i) p[i] *= inv;
  }
  __syncthreads();

  // ---- attn = probs @ V  (V transposed [d][m]; ping-pong, 2x unroll) ----
  const bf16_t* pV = vfT + (((long)b * NKV + hk) * DD + nloc) * SK + hi * 16;
  v8f acc[8] = {};
  v16bf vb0[8];
#pragma unroll
  for (int nf = 0; nf < 8; ++nf) vb0[nf] = pack16(pV + nf * 16 * SK, 8);
  for (int kc = 0; kc < 20; kc += 2) {            // key-chunks of 32
    v16bf vb1[8];
#pragma unroll
    for (int nf = 0; nf < 8; ++nf)
      vb1[nf] = pack16(pV + nf * 16 * SK + (kc + 1) * 32, 8);
    {
      const float* pr = sc + (wave * 16 + nloc) * SK + kc * 32 + hi * 16;
      v16bf a;
#pragma unroll
      for (int i = 0; i < 8; ++i) { a[i] = (bf16_t)pr[i]; a[8 + i] = (bf16_t)pr[8 + i]; }
#pragma unroll
      for (int nf = 0; nf < 8; ++nf)
        acc[nf] = __builtin_amdgcn_wmma_f32_16x16x32_bf16(
            false, a, false, vb0[nf], (short)0, acc[nf], false, false);
    }
    // next even chunk (dead over-read into adjacent workspace at kc=18)
#pragma unroll
    for (int nf = 0; nf < 8; ++nf)
      vb0[nf] = pack16(pV + nf * 16 * SK + (kc + 2) * 32, 8);
    {
      const float* pr = sc + (wave * 16 + nloc) * SK + (kc + 1) * 32 + hi * 16;
      v16bf a;
#pragma unroll
      for (int i = 0; i < 8; ++i) { a[i] = (bf16_t)pr[i]; a[8 + i] = (bf16_t)pr[8 + i]; }
#pragma unroll
      for (int nf = 0; nf < 8; ++nf)
        acc[nf] = __builtin_amdgcn_wmma_f32_16x16x32_bf16(
            false, a, false, vb1[nf], (short)0, acc[nf], false, false);
    }
  }
#pragma unroll
  for (int r = 0; r < 8; ++r) {
    const long t = (long)b * SS + qrow0 + r + 8 * hi;
#pragma unroll
    for (int nf = 0; nf < 8; ++nf)
      attnbf[t * HID + h * 128 + nf * 16 + nloc] = (bf16_t)acc[nf][r];
  }
}

// ---------------------------------------------------------------------------
// Kernel 7: output projection  (16384x2048) @ woT -> fp32 d_out
// ---------------------------------------------------------------------------
__global__ __launch_bounds__(256) void out_gemm(
    const bf16_t* __restrict__ attnbf, const bf16_t* __restrict__ woT,
    float* __restrict__ out) {
  const int lane = threadIdx.x & 31, wave = threadIdx.x >> 5;
  const int row0 = blockIdx.x * 128 + wave * 16;
  const int col0 = blockIdx.y * 128;
  v8f acc[8] = {};
  wave_gemm_strip(attnbf, HID, row0, woT, HID, col0, HID, acc);
  const int nloc = lane & 15, hi = lane >> 4;
#pragma unroll
  for (int r = 0; r < 8; ++r) {
    const long t = row0 + r + 8 * hi;
#pragma unroll
    for (int nf = 0; nf < 8; ++nf)
      out[t * HID + col0 + nf * 16 + nloc] = acc[nf][r];
  }
}

// ---------------------------------------------------------------------------
extern "C" void kernel_launch(void* const* d_in, const int* in_sizes, int n_in,
                              void* d_out, int out_size, void* d_ws, size_t ws_size,
                              hipStream_t stream) {
  (void)in_sizes; (void)n_in; (void)out_size; (void)ws_size;
  const float* x    = (const float*)d_in[0];
  const float* cosp = (const float*)d_in[1];
  const float* sinp = (const float*)d_in[2];
  const float* wq   = (const float*)d_in[3];
  const float* wk   = (const float*)d_in[4];
  const float* wv   = (const float*)d_in[5];
  const float* wo   = (const float*)d_in[6];
  const float* qnw  = (const float*)d_in[7];
  const float* knw  = (const float*)d_in[8];
  const float* ka   = (const float*)d_in[9];
  const float* kb   = (const float*)d_in[10];
  const float* iw1  = (const float*)d_in[11];
  const float* ib1  = (const float*)d_in[12];
  const float* iw2  = (const float*)d_in[13];
  const float* ib2  = (const float*)d_in[14];

  // workspace carve-up (256B aligned).  Order matters: every WMMA operand
  // buffer is followed by another region so tail prefetch over-reads stay
  // inside d_ws.
  char* ws = (char*)d_ws;
  size_t off = 0;
  auto carve = [&](size_t bytes) { char* p = ws + off; off += (bytes + 255) & ~(size_t)255; return p; };
  bf16_t* xbf    = (bf16_t*)carve((size_t)NTOK * HID * 2);   // reused as attnbf
  bf16_t* qbf    = (bf16_t*)carve((size_t)NTOK * HID * 2);
  bf16_t* kvwT   = (bf16_t*)carve((size_t)BB * 1024 * SS * 2);
  bf16_t* wqkvT  = (bf16_t*)carve((size_t)3072 * HID * 2);
  bf16_t* woT    = (bf16_t*)carve((size_t)HID * HID * 2);
  bf16_t* Pbf    = (bf16_t*)carve((size_t)SK * SS * 2);
  bf16_t* kpe    = (bf16_t*)carve((size_t)BB * SK * NKV * DD * 2);
  bf16_t* vfT    = (bf16_t*)carve((size_t)BB * NKV * DD * SK * 2);
  float*  iw     = (float*)carve((size_t)NTOK * 4);
  bf16_t* attnbf = xbf;   // x dead after qkv_gemm

  imp_x_kernel<<<NTOK, 256, 0, stream>>>(x, iw1, ib1, iw2, ib2, iw, xbf);
  build_wqkvT<<<(3072 * 2048) / 256, 256, 0, stream>>>(wq, wk, wv, wqkvT);
  build_woT<<<(2048 * 2048) / 256, 256, 0, stream>>>(wo, woT);
  build_P<<<(SK * SS) / 256, 256, 0, stream>>>(ka, kb, Pbf);

  qkv_gemm<<<dim3(NTOK / 128, 24), 256, 0, stream>>>(
      xbf, wqkvT, cosp, sinp, qnw, knw, iw, qbf, kvwT);
  sketch_gemm<<<dim3(SK / 128, 8, BB), 256, 0, stream>>>(
      Pbf, kvwT, cosp, sinp, kpe, vfT);
  attn_kernel<<<dim3(SS / 64, NH, BB), 128, 64 * SK * 4, stream>>>(
      qbf, kpe, vfT, attnbf);
  out_gemm<<<dim3(NTOK / 128, 16), 256, 0, stream>>>(attnbf, woT, (float*)d_out);
}
